// RSSMPriorRollout_67997922231002
// MI455X (gfx1250) — compile-verified, matrix operand order checked by reading, not run on hardware
//
#include <hip/hip_runtime.h>
#include <hip/hip_bf16.h>
#include <math.h>

// Problem dimensions (fixed by the reference)
#define B_   512
#define T_   64
#define S_   128
#define A_   32
#define H_   1024
#define R_   1024
#define KXA  160      // S_ + A_
#define N3R  3072     // 3 * R_

typedef __attribute__((ext_vector_type(16))) _Float16 v16h;
typedef __attribute__((ext_vector_type(8)))  _Float16 v8h;
typedef __attribute__((ext_vector_type(8)))  float    v8f;

// ---------------------------------------------------------------------------
// Fragment load: per-lane layout for 16-bit A/B 16x32 WMMA operands.
// Lane L: row = base_row + (L & 15); K half = (L >> 4) * 8.
// v16h elems 0..7  = M[row][k0 .. k0+7]      (VGPRs 0..3, packed pairs)
//      elems 8..15 = M[row][k0+16 .. k0+23]  (VGPRs 4..7)
// Both chunks are 16-byte aligned contiguous loads -> global_load_b128.
// ---------------------------------------------------------------------------
static __device__ __forceinline__ v16h load_frag(const _Float16* __restrict__ base,
                                                 int ld, int row, int k0) {
  const _Float16* p = base + (size_t)row * (size_t)ld + k0;
  v8h lo = *(const v8h*)(p);
  v8h hi = *(const v8h*)(p + 16);
  v16h r;
#pragma unroll
  for (int i = 0; i < 8; ++i) { r[i] = lo[i]; r[i + 8] = hi[i]; }
  return r;
}

// Epilogue store for one 16x16 f32 accumulator fragment.
// Elem j of lane L -> row (L>>4)*8 + j, col (L&15).
template <int MODE>
static __device__ __forceinline__ void store_frag(const v8f& c, int mbase, int nbase,
                                                  int lane,
                                                  const float* __restrict__ bias,
                                                  float* __restrict__ Cf,
                                                  _Float16* __restrict__ Ch,
                                                  int ldc) {
  const int col  = nbase + (lane & 15);
  const int row0 = mbase + ((lane >> 4) << 3);
  const float bv = bias ? bias[col] : 0.0f;
#pragma unroll
  for (int j = 0; j < 8; ++j) {
    float v = c[j] + bv;
    const size_t off = (size_t)(row0 + j) * (size_t)ldc + col;
    if (MODE == 0) {
      Cf[off] = v;                               // raw f32 (pre-gate / pre-sample)
    } else {
      float e = v > 0.0f ? v : expm1f(v);        // ELU
      Ch[off] = (_Float16)e;                     // f16 activation feeding next GEMM
    }
  }
}

// ---------------------------------------------------------------------------
// Generic WMMA GEMM:  C[M,N] = A[M,K] * B[K,N] + bias,  B given transposed
// (BT is [N,K] row-major so B-fragments load like A-fragments).
// Block = 128 threads = 4 waves (2x2); block tile 64x128.
// Wave tile 32x64: 2 A-frags x 4 B-frags -> 8 WMMA accumulators, so each
// K-chunk does 8 v_wmma per 12 global_load_b128 (vs 4 per 8 at 32x32),
// keeping the L2-resident weight stream ahead of the WMMA pipes.
// MODE 0: store f32 into Cf.  MODE 1: ELU -> f16 into Ch.
// ---------------------------------------------------------------------------
template <int MODE>
__global__ __launch_bounds__(128) void gemm_wmma_f16(
    const _Float16* __restrict__ A, int lda,
    const _Float16* __restrict__ BT, int ldb,
    const float* __restrict__ bias,
    float* __restrict__ Cf, _Float16* __restrict__ Ch,
    int K, int ldc) {
  const int lane = threadIdx.x & 31;
  const int wave = threadIdx.x >> 5;
  const int wm = wave >> 1;            // 2x2 wave grid inside the block
  const int wn = wave & 1;
  const int m0 = blockIdx.y * 64 + wm * 32;
  const int n0 = blockIdx.x * 128 + wn * 64;
  const int lrow = lane & 15;
  const int lk   = (lane >> 4) << 3;   // 0 or 8

  v8f c0[4] = {{}, {}, {}, {}};        // rows m0..m0+15
  v8f c1[4] = {{}, {}, {}, {}};        // rows m0+16..m0+31

  for (int kk = 0; kk < K; kk += 32) {
    if (kk + 64 < K) {  // speculative prefetch of the chunk after next
      __builtin_prefetch(A  + (size_t)(m0 + lrow) * lda + kk + 64, 0, 1);
      __builtin_prefetch(BT + (size_t)(n0 + lrow) * ldb + kk + 64, 0, 1);
      __builtin_prefetch(BT + (size_t)(n0 + 32 + lrow) * ldb + kk + 64, 0, 1);
    }
    v16h a0 = load_frag(A, lda, m0 + lrow,      kk + lk);
    v16h a1 = load_frag(A, lda, m0 + 16 + lrow, kk + lk);
#pragma unroll
    for (int i = 0; i < 4; ++i) {
      v16h b = load_frag(BT, ldb, n0 + 16 * i + lrow, kk + lk);
      c0[i] = __builtin_amdgcn_wmma_f32_16x16x32_f16(false, a0, false, b, (short)0, c0[i], false, false);
      c1[i] = __builtin_amdgcn_wmma_f32_16x16x32_f16(false, a1, false, b, (short)0, c1[i], false, false);
    }
  }

#pragma unroll
  for (int i = 0; i < 4; ++i) {
    store_frag<MODE>(c0[i], m0,      n0 + 16 * i, lane, bias, Cf, Ch, ldc);
    store_frag<MODE>(c1[i], m0 + 16, n0 + 16 * i, lane, bias, Cf, Ch, ldc);
  }
}

// ---------------------------------------------------------------------------
// One-time setup kernels (run every call -- deterministic, cheap vs. 0.5 TFLOP)
// ---------------------------------------------------------------------------
// dst[N][K] (f16) = transpose(src[K][N] (f32))
__global__ void k_convert_transpose(const float* __restrict__ src,
                                    _Float16* __restrict__ dst, int K, int N) {
  int idx = blockIdx.x * blockDim.x + threadIdx.x;
  if (idx >= K * N) return;
  int k = idx / N, n = idx - k * N;
  dst[(size_t)n * K + k] = (_Float16)src[idx];
}

__global__ void k_init_belief(const float* __restrict__ belief_in,
                              float* __restrict__ belief_f,
                              _Float16* __restrict__ belief_h) {
  int idx = blockIdx.x * blockDim.x + threadIdx.x;
  if (idx >= B_ * R_) return;
  float v = belief_in[idx];
  belief_f[idx] = v;
  belief_h[idx] = (_Float16)v;
}

__global__ void k_init_state(const float* __restrict__ prior_state,
                             _Float16* __restrict__ xa_h) {
  int idx = blockIdx.x * blockDim.x + threadIdx.x;
  if (idx >= B_ * S_) return;
  int b = idx / S_, j = idx - b * S_;
  xa_h[(size_t)b * KXA + j] = (_Float16)prior_state[idx];
}

// Inject action[:, t, :] into the concat buffer columns [S_, S_+A_)
__global__ void k_fill_action(const float* __restrict__ actions,
                              _Float16* __restrict__ xa_h, int t) {
  int idx = blockIdx.x * blockDim.x + threadIdx.x;
  if (idx >= B_ * A_) return;
  int b = idx / A_, a = idx - b * A_;
  xa_h[(size_t)b * KXA + S_ + a] = (_Float16)actions[((size_t)b * T_ + t) * A_ + a];
}

// ---------------------------------------------------------------------------
// GRU gating (torch GRUCell math) + belief bookkeeping + output writes
// ---------------------------------------------------------------------------
__global__ void k_gru(const float* __restrict__ gi, const float* __restrict__ gh,
                      float* __restrict__ belief_f, _Float16* __restrict__ belief_h,
                      float* __restrict__ out_beliefs,
                      float* __restrict__ out_prev_beliefs, int t) {
  int idx = blockIdx.x * blockDim.x + threadIdx.x;
  if (idx >= B_ * R_) return;
  int b = idx / R_, j = idx - b * R_;
  size_t gb = (size_t)b * N3R;
  float i_r = gi[gb + j], i_z = gi[gb + R_ + j], i_n = gi[gb + 2 * R_ + j];
  float h_r = gh[gb + j], h_z = gh[gb + R_ + j], h_n = gh[gb + 2 * R_ + j];
  float r = 1.0f / (1.0f + expf(-(i_r + h_r)));
  float z = 1.0f / (1.0f + expf(-(i_z + h_z)));
  float n = tanhf(i_n + r * h_n);
  float hp = belief_f[idx];
  float nb = (1.0f - z) * n + z * hp;
  size_t o = ((size_t)b * T_ + t) * R_ + j;
  out_prev_beliefs[o] = hp;
  out_beliefs[o] = nb;
  belief_f[idx] = nb;
  belief_h[idx] = (_Float16)nb;
}

// softplus/std, reparameterized sample, write means/stds/states, refresh state
// columns of the concat buffer for the next step.
__global__ void k_sample(const float* __restrict__ outf,
                         const float* __restrict__ noise,
                         float* __restrict__ out_means, float* __restrict__ out_stds,
                         float* __restrict__ out_states,
                         _Float16* __restrict__ xa_h, int t) {
  int idx = blockIdx.x * blockDim.x + threadIdx.x;
  if (idx >= B_ * S_) return;
  int b = idx / S_, j = idx - b * S_;
  float mean = outf[(size_t)b * (2 * S_) + j];
  float raw  = outf[(size_t)b * (2 * S_) + S_ + j];
  float sp   = raw > 20.0f ? raw : log1pf(expf(raw));  // softplus, stable
  float std  = sp + 0.1f;                              // MIN_STD
  float eps  = noise[((size_t)b * T_ + t) * S_ + j];
  float st   = mean + std * eps;
  size_t o = ((size_t)b * T_ + t) * S_ + j;
  out_means[o]  = mean;
  out_stds[o]   = std;
  out_states[o] = st;
  xa_h[(size_t)b * KXA + j] = (_Float16)st;
}

// ---------------------------------------------------------------------------
// Host-side orchestration
// ---------------------------------------------------------------------------
static inline size_t align256(size_t x) { return (x + 255) & ~(size_t)255; }

extern "C" void kernel_launch(void* const* d_in, const int* in_sizes, int n_in,
                              void* d_out, int out_size, void* d_ws, size_t ws_size,
                              hipStream_t stream) {
  (void)in_sizes; (void)n_in; (void)out_size; (void)ws_size;

  const float* prior_state = (const float*)d_in[0];
  const float* belief_in   = (const float*)d_in[1];
  const float* actions     = (const float*)d_in[2];
  const float* noise       = (const float*)d_in[3];
  const float* W_proj = (const float*)d_in[4];   // [160, 1024]
  const float* b_proj = (const float*)d_in[5];
  const float* W_ih   = (const float*)d_in[6];   // [1024, 3072]
  const float* b_ih   = (const float*)d_in[7];
  const float* W_hh   = (const float*)d_in[8];   // [1024, 3072]
  const float* b_hh   = (const float*)d_in[9];
  const float* W_h1   = (const float*)d_in[10];  // [1024, 1024]
  const float* b_h1   = (const float*)d_in[11];
  const float* W_out  = (const float*)d_in[12];  // [1024, 256]
  const float* b_out  = (const float*)d_in[13];

  // Output layout: means|stds|states|beliefs|prev_beliefs, each [B,T,...] flat
  float* out = (float*)d_out;
  const size_t nBTS = (size_t)B_ * T_ * S_;
  const size_t nBTR = (size_t)B_ * T_ * R_;
  float* out_means  = out;
  float* out_stds   = out + nBTS;
  float* out_states = out + 2 * nBTS;
  float* out_beliefs      = out + 3 * nBTS;
  float* out_prev_beliefs = out + 3 * nBTS + nBTR;

  // Workspace carve-up (all offsets 256B aligned)
  char* ws = (char*)d_ws;
  size_t off = 0;
  auto take = [&](size_t bytes) { char* p = ws + off; off += align256(bytes); return p; };
  _Float16* WT_proj = (_Float16*)take((size_t)H_  * KXA * 2);     // [1024][160]
  _Float16* WT_ih   = (_Float16*)take((size_t)N3R * H_  * 2);     // [3072][1024]
  _Float16* WT_hh   = (_Float16*)take((size_t)N3R * R_  * 2);     // [3072][1024]
  _Float16* WT_h1   = (_Float16*)take((size_t)H_  * R_  * 2);     // [1024][1024]
  _Float16* WT_out  = (_Float16*)take((size_t)(2 * S_) * H_ * 2); // [256][1024]
  _Float16* xa_h    = (_Float16*)take((size_t)B_ * KXA * 2);      // concat [state|action]
  _Float16* x_h     = (_Float16*)take((size_t)B_ * H_ * 2);
  float*    belief_f= (float*)   take((size_t)B_ * R_ * 4);
  _Float16* belief_h= (_Float16*)take((size_t)B_ * R_ * 2);
  float*    gi_f    = (float*)   take((size_t)B_ * N3R * 4);
  float*    gh_f    = (float*)   take((size_t)B_ * N3R * 4);
  _Float16* h_h     = (_Float16*)take((size_t)B_ * H_ * 2);
  float*    outp_f  = (float*)   take((size_t)B_ * 2 * S_ * 4);

  const int TPB = 256;
  auto blocks = [&](size_t n) { return (unsigned)((n + TPB - 1) / TPB); };

  // --- weight conversion + transpose (f32 [K][N] -> f16 [N][K]) ---
  k_convert_transpose<<<blocks((size_t)KXA * H_),  TPB, 0, stream>>>(W_proj, WT_proj, KXA, H_);
  k_convert_transpose<<<blocks((size_t)H_ * N3R),  TPB, 0, stream>>>(W_ih,   WT_ih,   H_,  N3R);
  k_convert_transpose<<<blocks((size_t)R_ * N3R),  TPB, 0, stream>>>(W_hh,   WT_hh,   R_,  N3R);
  k_convert_transpose<<<blocks((size_t)R_ * H_),   TPB, 0, stream>>>(W_h1,   WT_h1,   R_,  H_);
  k_convert_transpose<<<blocks((size_t)H_ * 2*S_), TPB, 0, stream>>>(W_out,  WT_out,  H_,  2 * S_);

  // --- recurrent state init ---
  k_init_belief<<<blocks((size_t)B_ * R_), TPB, 0, stream>>>(belief_in, belief_f, belief_h);
  k_init_state <<<blocks((size_t)B_ * S_), TPB, 0, stream>>>(prior_state, xa_h);

  dim3 blk(128);
  dim3 g_x (H_  / 128, B_ / 64);       // 8 x 8
  dim3 g_g (N3R / 128, B_ / 64);       // 24 x 8
  dim3 g_h1(H_  / 128, B_ / 64);       // 8 x 8
  dim3 g_o ((2 * S_) / 128, B_ / 64);  // 2 x 8

  for (int t = 0; t < T_; ++t) {
    // concat buffer <- action[:, t, :]
    k_fill_action<<<blocks((size_t)B_ * A_), TPB, 0, stream>>>(actions, xa_h, t);

    // x = ELU([state|action] @ W_proj + b_proj)           (f16 out)
    gemm_wmma_f16<1><<<g_x, blk, 0, stream>>>(xa_h, KXA, WT_proj, KXA, b_proj,
                                              nullptr, x_h, KXA, H_);
    // gi = x @ W_ih + b_ih ; gh = belief @ W_hh + b_hh     (f32 out)
    gemm_wmma_f16<0><<<g_g, blk, 0, stream>>>(x_h, H_, WT_ih, H_, b_ih,
                                              gi_f, nullptr, H_, N3R);
    gemm_wmma_f16<0><<<g_g, blk, 0, stream>>>(belief_h, R_, WT_hh, R_, b_hh,
                                              gh_f, nullptr, R_, N3R);
    // GRU gate math; writes beliefs / prev_beliefs outputs, updates belief bufs
    k_gru<<<blocks((size_t)B_ * R_), TPB, 0, stream>>>(gi_f, gh_f, belief_f, belief_h,
                                                       out_beliefs, out_prev_beliefs, t);
    // h = ELU(belief' @ W_h1 + b_h1)                       (f16 out)
    gemm_wmma_f16<1><<<g_h1, blk, 0, stream>>>(belief_h, R_, WT_h1, R_, b_h1,
                                               nullptr, h_h, R_, H_);
    // [mean|raw_std] = h @ W_out + b_out                   (f32 out)
    gemm_wmma_f16<0><<<g_o, blk, 0, stream>>>(h_h, H_, WT_out, H_, b_out,
                                              outp_f, nullptr, H_, 2 * S_);
    // std = softplus + MIN_STD; state = mean + std*eps; refresh concat buffer
    k_sample<<<blocks((size_t)B_ * S_), TPB, 0, stream>>>(outp_f, noise, out_means,
                                                          out_stds, out_states, xa_h, t);
  }
}
